// MyModel_87522843558912
// MI455X (gfx1250) — compile-verified, matrix operand order checked by reading, not run on hardware
//
#include <hip/hip_runtime.h>
#include <hip/hip_bf16.h>
#include <math.h>

#define EMB   2048
#define HH    2048
#define OUTP  10
#define BB    8
#define TT    512

typedef __attribute__((ext_vector_type(16))) __bf16       v16bf;
typedef __attribute__((ext_vector_type(8)))  float        v8f;
typedef __attribute__((ext_vector_type(4)))  unsigned int u32x4;

union Frag { u32x4 q[2]; v16bf v; };

__device__ __forceinline__ float sigm(float x) { return 1.f / (1.f + __expf(-x)); }
__device__ __forceinline__ float bf2f(__bf16 x) { return (float)x; }
__device__ __forceinline__ v8f vzero8() { v8f z = {0.f,0.f,0.f,0.f,0.f,0.f,0.f,0.f}; return z; }

__device__ __forceinline__ v8f wmma_bf16(const Frag& a, const Frag& b, v8f c) {
  // D = A(16x32 bf16) x B(32x16 bf16) + C(16x16 f32)
  return __builtin_amdgcn_wmma_f32_16x16x32_bf16(false, a.v, false, b.v, (short)0, c, false, false);
}

// A fragment: 16x32 bf16, row = lane&15, K-halves selected by lane<16 per ISA layout.
__device__ __forceinline__ void loadA(Frag& a, const __bf16* __restrict__ arow, int kk, int koff) {
  a.q[0] = *(const u32x4*)(arow + kk * 32 + koff);
  a.q[1] = *(const u32x4*)(arow + kk * 32 + 16 + koff);
}
// B fragment: packed fragment-major, 32B contiguous per lane.
__device__ __forceinline__ void loadB(Frag& b, const __bf16* __restrict__ Bp, size_t frag, size_t lo) {
  const __bf16* p = Bp + ((frag << 9) + lo);
  b.q[0] = *(const u32x4*)p;
  b.q[1] = *(const u32x4*)(p + 8);
}

// ---------------------------------------------------------------- zero init
__global__ void k_zero(unsigned int* __restrict__ p, long long n) {
  long long i = (long long)blockIdx.x * blockDim.x + threadIdx.x;
  if (i < n) p[i] = 0u;
}

// ------------------------------------------------- embedding lookup -> bf16
__global__ void k_embed(const int* __restrict__ inp, const float* __restrict__ emb,
                        __bf16* __restrict__ x) {
  long long i = (long long)blockIdx.x * blockDim.x + threadIdx.x;   // over B*T*EMB
  if (i >= (long long)BB * TT * EMB) return;
  long long bt = i >> 11;             // / EMB
  int e = (int)(i & (EMB - 1));
  int tok = inp[bt]; if (tok < 0) tok = 0;
  x[i] = (__bf16)emb[(long long)tok * EMB + e];
}

// -------- pack fp32 weight [K x N] into bf16 WMMA B-fragment-major layout.
// Fragment f = ntile*Kt + ktile covers rows k0..k0+31, cols n0..n0+15.
// Lane l holds col n0+(l&15); halfs[0..7]=B[k0+(l<16?0:8)+i][n], halfs[8..15]=B[k0+16+(l<16?0:8)+i][n].
__global__ void k_pack(const float* __restrict__ W, __bf16* __restrict__ P, int K, int N) {
  long long gid = (long long)blockIdx.x * blockDim.x + threadIdx.x;  // (N/16)*(K/32)*32 threads
  long long total = ((long long)N * K) >> 4;
  if (gid >= total) return;
  int lane = (int)(gid & 31);
  long long frag = gid >> 5;
  int Kt = K >> 5;
  int ktile = (int)(frag % Kt);
  long long ntile = frag / Kt;
  int n = (int)ntile * 16 + (lane & 15);
  int koff = (lane < 16) ? 0 : 8;
  int k0 = ktile * 32;
  union { __bf16 h[16]; u32x4 q[2]; } tmp;
  #pragma unroll
  for (int i = 0; i < 8; ++i) tmp.h[i]     = (__bf16)W[(long long)(k0 + koff + i) * N + n];
  #pragma unroll
  for (int i = 0; i < 8; ++i) tmp.h[8 + i] = (__bf16)W[(long long)(k0 + 16 + koff + i) * N + n];
  u32x4* dst = (u32x4*)(P + ((frag * 32 + lane) << 4));
  dst[0] = tmp.q[0]; dst[1] = tmp.q[1];
}

// ------------------------- input projection: xp = x @ W + bias  (bf16 out)
// One wave -> 16(M) x 64(N) tile; two-stage software pipeline over K chunks of 32.
__global__ void __launch_bounds__(256)
k_gemm_xp(const __bf16* __restrict__ A, const __bf16* __restrict__ Bp,
          const float* __restrict__ bias, __bf16* __restrict__ Cout,
          int M, int N, int K) {
  int tid = threadIdx.x, lane = tid & 31;
  int wid = blockIdx.x * (blockDim.x >> 5) + (tid >> 5);
  int n4 = N >> 6;
  if (wid >= (M >> 4) * n4) return;
  int mt = wid / n4, nt4 = wid % n4;
  int Kt = K >> 5;
  const __bf16* arow = A + (long long)(mt * 16 + (lane & 15)) * K;
  const int koff = (lane < 16) ? 0 : 8;
  v8f acc[4];
  #pragma unroll
  for (int s = 0; s < 4; ++s) acc[s] = vzero8();
  const size_t fb = (size_t)nt4 * 4 * Kt;
  const size_t lo = (size_t)lane << 4;

  Frag a0, a1;
  Frag b0[4], b1[4];
  loadA(a0, arow, 0, koff);
  #pragma unroll
  for (int s = 0; s < 4; ++s) loadB(b0[s], Bp, fb + (size_t)s * Kt + 0, lo);

  for (int kk = 0; kk < Kt; kk += 2) {
    // stage 1 loads (kk+1) in flight while stage 0 computes
    loadA(a1, arow, kk + 1, koff);
    #pragma unroll
    for (int s = 0; s < 4; ++s) loadB(b1[s], Bp, fb + (size_t)s * Kt + kk + 1, lo);
    if (kk + 4 < Kt)
      __builtin_prefetch((const void*)(Bp + (((fb + kk + 4) << 9) + lo)), 0, 3);
    #pragma unroll
    for (int s = 0; s < 4; ++s) acc[s] = wmma_bf16(a0, b0[s], acc[s]);
    if (kk + 2 < Kt) {
      loadA(a0, arow, kk + 2, koff);
      #pragma unroll
      for (int s = 0; s < 4; ++s) loadB(b0[s], Bp, fb + (size_t)s * Kt + kk + 2, lo);
    }
    #pragma unroll
    for (int s = 0; s < 4; ++s) acc[s] = wmma_bf16(a1, b1[s], acc[s]);
  }

  // C layout: VGPR r -> row m0+r (lanes 0-15) / m0+8+r (lanes 16-31), col = n0 + (lane&15)
  int m0 = mt * 16, n0 = nt4 * 64;
  int cm = m0 + ((lane < 16) ? 0 : 8);
  #pragma unroll
  for (int s = 0; s < 4; ++s) {
    int cn = n0 + s * 16 + (lane & 15);
    float bs = bias[cn];
    #pragma unroll
    for (int r = 0; r < 8; ++r)
      Cout[(long long)(cm + r) * N + cn] = (__bf16)(acc[s][r] + bs);
  }
}

// ------------------------------------------ GRU recurrent step (both dirs)
// 256 waves: dir = wid>>7 (0 fwd, 1 bwd), jt = wid&127 -> 16-col slice of h.
__global__ void __launch_bounds__(256)
k_step_gru(int t, const int* __restrict__ inp,
           const __bf16* __restrict__ xpf, const __bf16* __restrict__ xpb,
           const __bf16* __restrict__ Uf,  const __bf16* __restrict__ Ub,
           const float* __restrict__ brf,  const float* __restrict__ brb,
           const __bf16* __restrict__ hinf, __bf16* __restrict__ houtf,
           const __bf16* __restrict__ hinb, __bf16* __restrict__ houtb,
           float* __restrict__ ysf, float* __restrict__ ysb) {
  const int tid = threadIdx.x, lane = tid & 31;
  const int wid = blockIdx.x * (blockDim.x >> 5) + (tid >> 5);
  const int dir = wid >> 7, jt = wid & 127;
  const int te = dir ? (TT - 1 - t) : t;
  const __bf16* xp = dir ? xpb : xpf;
  const __bf16* U  = dir ? Ub  : Uf;
  const float*  br = dir ? brb : brf;
  const __bf16* hin = dir ? hinb : hinf;
  __bf16* hout = dir ? houtb : houtf;
  float* ys = dir ? ysb : ysf;

  const __bf16* arow = hin + (size_t)(lane & 15) * HH;
  const int koff = (lane < 16) ? 0 : 8;
  v8f acc0 = vzero8(), acc1 = vzero8(), acc2 = vzero8();
  const int Kt = HH / 32;
  const size_t lo = (size_t)lane << 4;
  const size_t f0 = ((size_t)(0 * 128 + jt)) * Kt;
  const size_t f1 = ((size_t)(1 * 128 + jt)) * Kt;
  const size_t f2 = ((size_t)(2 * 128 + jt)) * Kt;

  Frag a0, a1, b00, b01, b02, b10, b11, b12;
  loadA(a0, arow, 0, koff);
  loadB(b00, U, f0 + 0, lo);
  loadB(b01, U, f1 + 0, lo);
  loadB(b02, U, f2 + 0, lo);

  for (int kk = 0; kk < Kt; kk += 2) {
    loadA(a1, arow, kk + 1, koff);
    loadB(b10, U, f0 + kk + 1, lo);
    loadB(b11, U, f1 + kk + 1, lo);
    loadB(b12, U, f2 + kk + 1, lo);
    acc0 = wmma_bf16(a0, b00, acc0);
    acc1 = wmma_bf16(a0, b01, acc1);
    acc2 = wmma_bf16(a0, b02, acc2);
    if (kk + 2 < Kt) {
      loadA(a0, arow, kk + 2, koff);
      loadB(b00, U, f0 + kk + 2, lo);
      loadB(b01, U, f1 + kk + 2, lo);
      loadB(b02, U, f2 + kk + 2, lo);
    }
    acc0 = wmma_bf16(a1, b10, acc0);
    acc1 = wmma_bf16(a1, b11, acc1);
    acc2 = wmma_bf16(a1, b12, acc2);
  }

  if (lane < 16) {
    const int j = jt * 16 + lane;
    #pragma unroll
    for (int b = 0; b < 8; ++b) {
      long long xb = ((long long)b * TT + te) * (3 * HH);
      float rz = acc0[b] + br[j];
      float rr = acc1[b] + br[HH + j];
      float rh = acc2[b] + br[2 * HH + j];
      float z  = sigm(bf2f(xp[xb + j])           + rz);
      float r  = sigm(bf2f(xp[xb + HH + j])      + rr);
      float hh = tanhf(bf2f(xp[xb + 2 * HH + j]) + r * rh);
      float hp = bf2f(hin[(size_t)b * HH + j]);
      float hn = z * hp + (1.f - z) * hh;
      if (inp[b * TT + te] < 0) hn = hp;
      hout[(size_t)b * HH + j] = (__bf16)hn;
      ys[((long long)b * TT + te) * HH + j] = hn;
    }
  } else {
    const int j = jt * 16 + (lane - 16);
    #pragma unroll
    for (int b = 8; b < 16; ++b) hout[(size_t)b * HH + j] = (__bf16)0.f;  // keep pad rows zero
  }
}

// ----------------------------------------- LSTM recurrent step (both dirs)
__global__ void __launch_bounds__(256)
k_step_lstm(int t, const int* __restrict__ inp,
            const __bf16* __restrict__ xpf, const __bf16* __restrict__ xpb,
            const __bf16* __restrict__ Uf,  const __bf16* __restrict__ Ub,
            const __bf16* __restrict__ hinf, __bf16* __restrict__ houtf,
            const __bf16* __restrict__ hinb, __bf16* __restrict__ houtb,
            float* __restrict__ cf, float* __restrict__ cb,
            float* __restrict__ ysf, float* __restrict__ ysb) {
  const int tid = threadIdx.x, lane = tid & 31;
  const int wid = blockIdx.x * (blockDim.x >> 5) + (tid >> 5);
  const int dir = wid >> 7, jt = wid & 127;
  const int te = dir ? (TT - 1 - t) : t;
  const __bf16* xp = dir ? xpb : xpf;
  const __bf16* U  = dir ? Ub  : Uf;
  const __bf16* hin = dir ? hinb : hinf;
  __bf16* hout = dir ? houtb : houtf;
  float* cs = dir ? cb : cf;
  float* ys = dir ? ysb : ysf;

  const __bf16* arow = hin + (size_t)(lane & 15) * HH;
  const int koff = (lane < 16) ? 0 : 8;
  v8f acc0 = vzero8(), acc1 = vzero8(), acc2 = vzero8(), acc3 = vzero8();
  const int Kt = HH / 32;
  const size_t lo = (size_t)lane << 4;
  const size_t f0 = ((size_t)(0 * 128 + jt)) * Kt;
  const size_t f1 = ((size_t)(1 * 128 + jt)) * Kt;
  const size_t f2 = ((size_t)(2 * 128 + jt)) * Kt;
  const size_t f3 = ((size_t)(3 * 128 + jt)) * Kt;

  Frag a0, a1, b00, b01, b02, b03, b10, b11, b12, b13;
  loadA(a0, arow, 0, koff);
  loadB(b00, U, f0 + 0, lo);
  loadB(b01, U, f1 + 0, lo);
  loadB(b02, U, f2 + 0, lo);
  loadB(b03, U, f3 + 0, lo);

  for (int kk = 0; kk < Kt; kk += 2) {
    loadA(a1, arow, kk + 1, koff);
    loadB(b10, U, f0 + kk + 1, lo);
    loadB(b11, U, f1 + kk + 1, lo);
    loadB(b12, U, f2 + kk + 1, lo);
    loadB(b13, U, f3 + kk + 1, lo);
    acc0 = wmma_bf16(a0, b00, acc0);
    acc1 = wmma_bf16(a0, b01, acc1);
    acc2 = wmma_bf16(a0, b02, acc2);
    acc3 = wmma_bf16(a0, b03, acc3);
    if (kk + 2 < Kt) {
      loadA(a0, arow, kk + 2, koff);
      loadB(b00, U, f0 + kk + 2, lo);
      loadB(b01, U, f1 + kk + 2, lo);
      loadB(b02, U, f2 + kk + 2, lo);
      loadB(b03, U, f3 + kk + 2, lo);
    }
    acc0 = wmma_bf16(a1, b10, acc0);
    acc1 = wmma_bf16(a1, b11, acc1);
    acc2 = wmma_bf16(a1, b12, acc2);
    acc3 = wmma_bf16(a1, b13, acc3);
  }

  if (lane < 16) {
    const int j = jt * 16 + lane;
    #pragma unroll
    for (int b = 0; b < 8; ++b) {
      long long xb = ((long long)b * TT + te) * (4 * HH);
      float gi = bf2f(xp[xb + j])          + acc0[b];
      float gf = bf2f(xp[xb + HH + j])     + acc1[b];
      float gc = bf2f(xp[xb + 2 * HH + j]) + acc2[b];
      float go = bf2f(xp[xb + 3 * HH + j]) + acc3[b];
      float i_ = sigm(gi), ff = sigm(gf), o_ = sigm(go);
      float cp = cs[(size_t)b * HH + j];
      float cn = ff * cp + i_ * tanhf(gc);
      float hp = bf2f(hin[(size_t)b * HH + j]);
      float hn = o_ * tanhf(cn);
      if (inp[b * TT + te] < 0) { hn = hp; cn = cp; }
      cs[(size_t)b * HH + j] = cn;
      hout[(size_t)b * HH + j] = (__bf16)hn;
      ys[((long long)b * TT + te) * HH + j] = hn;
    }
  } else {
    const int j = jt * 16 + (lane - 16);
    #pragma unroll
    for (int b = 8; b < 16; ++b) hout[(size_t)b * HH + j] = (__bf16)0.f;
  }
}

// -------------------------- out = (ys_gf+ys_gb+ys_lf+ys_lb) @ Wd + bd
__global__ void __launch_bounds__(256)
k_out(const float* __restrict__ y0, const float* __restrict__ y1,
      const float* __restrict__ y2, const float* __restrict__ y3,
      const float* __restrict__ Wd, const float* __restrict__ bd,
      float* __restrict__ out) {
  const int row = blockIdx.x;        // 0 .. B*T-1
  const int tid = threadIdx.x;
  __shared__ float sacc[OUTP];
  if (tid < OUTP) sacc[tid] = 0.f;
  __syncthreads();
  float a[OUTP];
  #pragma unroll
  for (int o = 0; o < OUTP; ++o) a[o] = 0.f;
  const long long base = (long long)row * HH;
  for (int h = tid; h < HH; h += blockDim.x) {
    float c = y0[base + h] + y1[base + h] + y2[base + h] + y3[base + h];
    #pragma unroll
    for (int o = 0; o < OUTP; ++o) a[o] += c * Wd[h * OUTP + o];
  }
  #pragma unroll
  for (int o = 0; o < OUTP; ++o) atomicAdd(&sacc[o], a[o]);
  __syncthreads();
  if (tid < OUTP) out[(long long)row * OUTP + tid] = sacc[tid] + bd[tid];
}

// ---------------------------------------------------------------- launcher
extern "C" void kernel_launch(void* const* d_in, const int* in_sizes, int n_in,
                              void* d_out, int out_size, void* d_ws, size_t ws_size,
                              hipStream_t stream) {
  (void)in_sizes; (void)n_in; (void)out_size; (void)ws_size;
  const int*   inp   = (const int*)  d_in[0];
  const float* emb   = (const float*)d_in[1];
  const float* Wg_f  = (const float*)d_in[2];
  const float* Ug_f  = (const float*)d_in[3];
  const float* bgi_f = (const float*)d_in[4];
  const float* bgr_f = (const float*)d_in[5];
  const float* Wg_b  = (const float*)d_in[6];
  const float* Ug_b  = (const float*)d_in[7];
  const float* bgi_b = (const float*)d_in[8];
  const float* bgr_b = (const float*)d_in[9];
  const float* Wl_f  = (const float*)d_in[10];
  const float* Ul_f  = (const float*)d_in[11];
  const float* bl_f  = (const float*)d_in[12];
  const float* Wl_b  = (const float*)d_in[13];
  const float* Ul_b  = (const float*)d_in[14];
  const float* bl_b  = (const float*)d_in[15];
  const float* Wd    = (const float*)d_in[16];
  const float* bd    = (const float*)d_in[17];
  float* out = (float*)d_out;

  char* ws = (char*)d_ws;
  auto alloc = [&](size_t bytes) -> char* {
    char* p = ws; ws += (bytes + 255) & ~(size_t)255; return p;
  };
  const size_t G3 = (size_t)3 * HH, G4 = (size_t)4 * HH;
  __bf16* x      = (__bf16*)alloc((size_t)BB * TT * EMB * 2);
  __bf16* Wgf_p  = (__bf16*)alloc((size_t)EMB * G3 * 2);
  __bf16* Wgb_p  = (__bf16*)alloc((size_t)EMB * G3 * 2);
  __bf16* Wlf_p  = (__bf16*)alloc((size_t)EMB * G4 * 2);
  __bf16* Wlb_p  = (__bf16*)alloc((size_t)EMB * G4 * 2);
  __bf16* Ugf_p  = (__bf16*)alloc((size_t)HH * G3 * 2);
  __bf16* Ugb_p  = (__bf16*)alloc((size_t)HH * G3 * 2);
  __bf16* Ulf_p  = (__bf16*)alloc((size_t)HH * G4 * 2);
  __bf16* Ulb_p  = (__bf16*)alloc((size_t)HH * G4 * 2);
  __bf16* xp_gf  = (__bf16*)alloc((size_t)BB * TT * G3 * 2);
  __bf16* xp_gb  = (__bf16*)alloc((size_t)BB * TT * G3 * 2);
  __bf16* xp_lf  = (__bf16*)alloc((size_t)BB * TT * G4 * 2);
  __bf16* xp_lb  = (__bf16*)alloc((size_t)BB * TT * G4 * 2);
  float*  ys_gf  = (float*)alloc((size_t)BB * TT * HH * 4);
  float*  ys_gb  = (float*)alloc((size_t)BB * TT * HH * 4);
  float*  ys_lf  = (float*)alloc((size_t)BB * TT * HH * 4);
  float*  ys_lb  = (float*)alloc((size_t)BB * TT * HH * 4);
  __bf16* hbase  = (__bf16*)alloc((size_t)8 * 16 * HH * 2);  // 4 dirs x 2 phase buffers
  float*  cbase  = (float*)alloc((size_t)2 * 8 * HH * 4);    // LSTM c states

  __bf16* h_gf[2] = { hbase + 0 * 16 * HH, hbase + 1 * 16 * HH };
  __bf16* h_gb[2] = { hbase + 2 * 16 * HH, hbase + 3 * 16 * HH };
  __bf16* h_lf[2] = { hbase + 4 * 16 * HH, hbase + 5 * 16 * HH };
  __bf16* h_lb[2] = { hbase + 6 * 16 * HH, hbase + 7 * 16 * HH };
  float*  c_lf = cbase;
  float*  c_lb = cbase + 8 * HH;

  // 1) zero h / c state
  {
    long long nH = (long long)8 * 16 * HH / 2;   // bf16 region in u32 words
    long long nC = (long long)2 * 8 * HH;        // f32 region in u32 words
    k_zero<<<(int)((nH + 255) / 256), 256, 0, stream>>>((unsigned int*)hbase, nH);
    k_zero<<<(int)((nC + 255) / 256), 256, 0, stream>>>((unsigned int*)cbase, nC);
  }
  // 2) embedding -> bf16 x
  {
    long long n = (long long)BB * TT * EMB;
    k_embed<<<(int)((n + 255) / 256), 256, 0, stream>>>(inp, emb, x);
  }
  // 3) pack weights to bf16 fragment-major (all four U total 117MB bf16 -> L2-resident, 192MB L2)
  {
    int bl3 = (int)(((size_t)EMB * G3 / 16 + 255) / 256);
    int bl4 = (int)(((size_t)EMB * G4 / 16 + 255) / 256);
    k_pack<<<bl3, 256, 0, stream>>>(Wg_f, Wgf_p, EMB, (int)G3);
    k_pack<<<bl3, 256, 0, stream>>>(Wg_b, Wgb_p, EMB, (int)G3);
    k_pack<<<bl4, 256, 0, stream>>>(Wl_f, Wlf_p, EMB, (int)G4);
    k_pack<<<bl4, 256, 0, stream>>>(Wl_b, Wlb_p, EMB, (int)G4);
    k_pack<<<bl3, 256, 0, stream>>>(Ug_f, Ugf_p, HH, (int)G3);
    k_pack<<<bl3, 256, 0, stream>>>(Ug_b, Ugb_p, HH, (int)G3);
    k_pack<<<bl4, 256, 0, stream>>>(Ul_f, Ulf_p, HH, (int)G4);
    k_pack<<<bl4, 256, 0, stream>>>(Ul_b, Ulb_p, HH, (int)G4);
  }
  // 4) input projections (WMMA GEMMs), bias fused
  {
    const int M = BB * TT;
    int blocks3 = (M / 16) * ((int)G3 / 64) / 8;   // waves / 8
    int blocks4 = (M / 16) * ((int)G4 / 64) / 8;
    k_gemm_xp<<<blocks3, 256, 0, stream>>>(x, Wgf_p, bgi_f, xp_gf, M, (int)G3, EMB);
    k_gemm_xp<<<blocks3, 256, 0, stream>>>(x, Wgb_p, bgi_b, xp_gb, M, (int)G3, EMB);
    k_gemm_xp<<<blocks4, 256, 0, stream>>>(x, Wlf_p, bl_f,  xp_lf, M, (int)G4, EMB);
    k_gemm_xp<<<blocks4, 256, 0, stream>>>(x, Wlb_p, bl_b,  xp_lb, M, (int)G4, EMB);
  }
  // 5) sequential recurrence: 512 steps, double-buffered h state
  for (int t = 0; t < TT; ++t) {
    int pi = t & 1, po = pi ^ 1;
    k_step_gru<<<32, 256, 0, stream>>>(t, inp, xp_gf, xp_gb, Ugf_p, Ugb_p, bgr_f, bgr_b,
                                       h_gf[pi], h_gf[po], h_gb[pi], h_gb[po], ys_gf, ys_gb);
    k_step_lstm<<<32, 256, 0, stream>>>(t, inp, xp_lf, xp_lb, Ulf_p, Ulb_p,
                                        h_lf[pi], h_lf[po], h_lb[pi], h_lb[po],
                                        c_lf, c_lb, ys_lf, ys_lb);
  }
  // 6) combine + output projection
  k_out<<<BB * TT, 256, 0, stream>>>(ys_gf, ys_gb, ys_lf, ys_lb, Wd, bd, out);
}
